// Generator_5755256177028
// MI455X (gfx1250) — matrix operand order repaired and checked by reading, not compiled, not run on hardware
//
#include <hip/hip_runtime.h>
#include <hip/hip_bf16.h>
#include <math.h>

// ---------------- model constants ----------------
#define MM   5000
#define EE   128
#define DD   512
#define DII  2048
#define HH   8
#define DKK  64
#define CC   256
#define SS   32
#define BB   32
#define TT   64
#define TMM  63          // T-1
#define LPP  1954        // 1 + 63*31
#define LNN  2017        // 1 + 63*32
#define NTHR 512         // main kernel threads (16 wave32)

typedef __attribute__((ext_vector_type(16))) __bf16 v16bf;
typedef __attribute__((ext_vector_type(8)))  float  v8f;

// ---------------- small helpers ----------------
__device__ __forceinline__ unsigned hash_u32(unsigned x) {
  x ^= x >> 16; x *= 0x7feb352du;
  x ^= x >> 15; x *= 0x846ca68bu;
  x ^= x >> 16; return x;
}
__device__ __forceinline__ float u01f(unsigned x) {
  return (hash_u32(x) >> 8) * (1.0f / 16777216.0f) + 1.0e-7f;
}
__device__ __forceinline__ float gumbelf(unsigned x) {
  float u = u01f(x);
  return -logf(-logf(u));
}
__device__ __forceinline__ float wred_sum(float v) {
#pragma unroll
  for (int o = 16; o; o >>= 1) v += __shfl_xor(v, o, 32);
  return v;
}
__device__ __forceinline__ float wred_max(float v) {
#pragma unroll
  for (int o = 16; o; o >>= 1) v = fmaxf(v, __shfl_xor(v, o, 32));
  return v;
}

// ---------------- WMMA fragment loaders (16x16x32 bf16) ----------------
// A: 16x32 (MxK).  lane m = lane&15, half = lane>>4.
// element e of v16bf pairs into VGPR e/2; VGPR j holds
// K = half*16 + j*2 (+1)
__device__ __forceinline__ v16bf load_a_f32(const float* A, int lda,
                                            int rowBase, int kBase, int lane) {
  int m = rowBase + (lane & 15);
  int half = lane >> 4;
  const float* row = A + (size_t)m * lda + kBase;
  v16bf a;
#pragma unroll
  for (int j = 0; j < 8; ++j) {
    int k0 = (half << 4) + (j << 1);
    a[2 * j]     = (__bf16)row[k0];
    a[2 * j + 1] = (__bf16)row[k0 + 1];
  }
  return a;
}
// A with optional row-gather + clamp (for precompute GEMMs)
__device__ __forceinline__ v16bf load_a_gather(const float* A, int lda,
                                               const int* rowIdx, int Mrows,
                                               int rowBase, int kBase, int lane) {
  int m = rowBase + (lane & 15);
  if (m >= Mrows) m = Mrows - 1;
  int r = rowIdx ? rowIdx[m] : m;
  int half = lane >> 4;
  const float* row = A + (size_t)r * lda + kBase;
  v16bf a;
#pragma unroll
  for (int j = 0; j < 8; ++j) {
    int k0 = (half << 4) + (j << 1);
    a[2 * j]     = (__bf16)row[k0];
    a[2 * j + 1] = (__bf16)row[k0 + 1];
  }
  return a;
}
// B fragments come from the PRE-SWIZZLED weight image: for each
// (k-block kb = k/32, n-tile nt = n/16) the 32 lanes' 16 bf16 values are
// stored contiguously: Wsw[((kb*tilesN + nt)*32 + lane)*16 + e],
// where lane = (kl>>4)*16 + (n&15) and e = kl&15 (kl = k&31).
// -> a single aligned 32-byte vector load per fragment.
__device__ __forceinline__ v16bf load_b_sw(const v16bf* Wsw, int tilesN,
                                           int kk, int colBase, int lane) {
  return Wsw[((size_t)(kk >> 5) * tilesN + (colBase >> 4)) * 32 + lane];
}

// GEMM: Out(32 x N) = A(32 x K, f32, lda) @ Wsw(K x N swizzled bf16).
// Runs on NTHR threads (16 waves), tiles = 2 * N/16, raw accum store.
__device__ void gemm32(const float* A, int lda, const v16bf* Wsw, int K, int N,
                       float* Out, int ldo, int tid) {
  int w = tid >> 5, lane = tid & 31;
  int tilesN = N >> 4;
  int nTiles = tilesN << 1;
  for (int t = w; t < nTiles; t += 16) {
    int rowBase = (t & 1) << 4;
    int colBase = (t >> 1) << 4;
    v8f c = {};
    for (int kk = 0; kk < K; kk += 32) {
      if (kk + 32 < K)
        __builtin_prefetch(Wsw + ((size_t)((kk + 32) >> 5) * tilesN +
                                  (colBase >> 4)) * 32 + lane, 0, 1);
      v16bf a = load_a_f32(A, lda, rowBase, kk, lane);
      v16bf b = load_b_sw(Wsw, tilesN, kk, colBase, lane);
      c = __builtin_amdgcn_wmma_f32_16x16x32_bf16(false, a, false, b,
                                                  (short)0, c, false, false);
    }
    int n = colBase + (lane & 15);
#pragma unroll
    for (int r = 0; r < 8; ++r) {
      int m = rowBase + ((lane >> 4) << 3) + r;
      Out[(size_t)m * ldo + n] = c[r];
    }
  }
}

// ---------------- precompute kernels ----------------
// fp32 (K x N) row-major -> bf16 WMMA-B-fragment-swizzled image
__global__ void k_swizzle_w(const float* __restrict__ src,
                            __bf16* __restrict__ dst, int K, int N) {
  int total = K * N;
  int st = gridDim.x * blockDim.x;
  for (int idx = blockIdx.x * blockDim.x + threadIdx.x; idx < total; idx += st) {
    int k = idx / N, n = idx - k * N;
    int kb = k >> 5, kl = k & 31;
    int half = kl >> 4, e = kl & 15;
    int nt = n >> 4, nl = n & 15;
    int lane = (half << 4) | nl;
    size_t o = (((size_t)kb * (N >> 4) + nt) * 32 + lane) * 16 + e;
    dst[o] = (__bf16)src[idx];
  }
}

__global__ void k_cand_init(int* cand) {
  int c = threadIdx.x;
  cand[c] = (int)(hash_u32(0x9E3779B9u + c * 2654435761u) % MM);
}

// Generic Out(M x N) = gather(A)(M x K) @ Wsw(K x N swizzled bf16);
// one 16x16 tile per wave.
__global__ void k_gemm_pre(const float* A, const int* rowIdx, int Mrows, int K,
                           const v16bf* Wsw, int N, float* Out) {
  int wave = (blockIdx.x * blockDim.x + threadIdx.x) >> 5;
  int lane = threadIdx.x & 31;
  int tilesN = N >> 4;
  int tilesM = (Mrows + 15) >> 4;
  if (wave >= tilesM * tilesN) return;           // wave-uniform exit
  int rowBase = (wave / tilesN) << 4;
  int colBase = (wave % tilesN) << 4;
  v8f c = {};
  for (int kk = 0; kk < K; kk += 32) {
    v16bf a = load_a_gather(A, K, rowIdx, Mrows, rowBase, kk, lane);
    v16bf b = load_b_sw(Wsw, tilesN, kk, colBase, lane);
    c = __builtin_amdgcn_wmma_f32_16x16x32_bf16(false, a, false, b,
                                                (short)0, c, false, false);
  }
  int n = colBase + (lane & 15);
#pragma unroll
  for (int r = 0; r < 8; ++r) {
    int m = rowBase + ((lane >> 4) << 3) + r;
    if (m < Mrows) Out[(size_t)m * N + n] = c[r];
  }
}

// p[m][c] = softmax_c( sum_e lrelu(Acand[c][e]+Bm[m][e]+s1_b[e]) * s2_w[e] + s2_b )
__global__ void k_pscore(const float* __restrict__ Acand,
                         const float* __restrict__ Bm,
                         const float* __restrict__ s1_b,
                         const float* __restrict__ s2_w,
                         const float* __restrict__ s2_b,
                         float* __restrict__ p) {
  int m = blockIdx.x, c = threadIdx.x;
  const float* ac = Acand + (size_t)c * EE;
  const float* bm = Bm + (size_t)m * EE;
  float acc = s2_b[0];
  for (int e = 0; e < EE; ++e) {
    float t = ac[e] + bm[e] + s1_b[e];
    t = (t >= 0.f) ? t : 0.01f * t;
    acc += t * s2_w[e];
  }
  __shared__ float red[CC];
  red[c] = acc; __syncthreads();
  for (int st = CC / 2; st; st >>= 1) {
    if (c < st) red[c] = fmaxf(red[c], red[c + st]);
    __syncthreads();
  }
  float mx = red[0]; __syncthreads();
  float e = expf(acc - mx);
  red[c] = e; __syncthreads();
  for (int st = CC / 2; st; st >>= 1) {
    if (c < st) red[c] += red[c + st];
    __syncthreads();
  }
  p[(size_t)m * CC + c] = e / red[0];
}

// per-m top-S via log(p)+gumbel, one wave per m
__global__ void k_topk(const float* __restrict__ p, const int* __restrict__ cand,
                       int* __restrict__ nlist, float* __restrict__ nprob) {
  int m = blockIdx.x, lane = threadIdx.x;
  __shared__ float sc[CC];
  __shared__ float pv[CC];
  for (int c = lane; c < CC; c += 32) {
    float q = p[(size_t)m * CC + c];
    pv[c] = q;
    sc[c] = logf(fmaxf(q, 1e-30f)) + gumbelf((unsigned)(m * 977 + c) * 0x9E3779B9u);
  }
  __syncthreads();
  for (int s = 0; s < SS; ++s) {
    float best = -3.0e38f; int bi = 0;
    for (int c = lane; c < CC; c += 32)
      if (sc[c] > best) { best = sc[c]; bi = c; }
#pragma unroll
    for (int o = 16; o; o >>= 1) {
      float ov = __shfl_xor(best, o, 32);
      int   oi = __shfl_xor(bi, o, 32);
      if (ov > best || (ov == best && oi < bi)) { best = ov; bi = oi; }
    }
    if (lane == 0) {
      nlist[(size_t)m * SS + s] = cand[bi];
      nprob[(size_t)m * SS + s] = pv[bi];
      sc[bi] = -3.0e38f;
    }
    __syncthreads();
  }
}

// ---------------- main persistent sequential kernel ----------------
struct MainParams {
  const int*   marker; const float* timed;  const float* maskd;
  const float* emb;    const float* te_w;   const float* te_b;
  const float* el_b;   const float* wo_b;
  const float* ln1_s;  const float* ln1_b;
  const float* ff1_b;  const float* ff2_b;
  const float* ln2_s;  const float* ln2_b;
  const float* tl_w;   const float* tl_b;
  const float* ml_w;   const float* ml_b;
  const v16bf* el_w;   const v16bf* wq; const v16bf* wk; const v16bf* wv;
  const v16bf* wo;     const v16bf* ff1; const v16bf* ff2;
  const int*   nlist;  const float* nprob;
  float* kc; float* vc; float* g_q; float* g_ctx; float* g_h;
  float* prob_list; int* cand_list; float* nrec;
  float* out;
};

__device__ void ln_rows(float* X, const float* s, const float* bb,
                        int w, int lane) {
  for (int r = w; r < BB; r += 16) {
    float acc = 0.f;
    for (int k = lane; k < DD; k += 32) acc += X[r * DD + k];
    float mean = wred_sum(acc) / (float)DD;
    float v = 0.f;
    for (int k = lane; k < DD; k += 32) {
      float d = X[r * DD + k] - mean; v += d * d;
    }
    float var = wred_sum(v) / (float)DD;
    float inv = rsqrtf(var + 1e-5f);
    for (int k = lane; k < DD; k += 32)
      X[r * DD + k] = (X[r * DD + k] - mean) * inv * s[k] + bb[k];
  }
}

__global__ __launch_bounds__(NTHR) void k_main(MainParams P) {
  const int tid = threadIdx.x;
  const int w = tid >> 5, lane = tid & 31;

  __shared__ float sA[BB][DD];        // 64 KB
  __shared__ float sB[BB][DD];        // 64 KB
  __shared__ float sScore[16][64];    // per-wave attention scores
  __shared__ float sMW[BB][SS];
  __shared__ float sDot[BB];          // inten . ml_w[E:]
  __shared__ float sTime[BB];
  __shared__ int   sMark[BB];
  __shared__ int   sChosen[BB];

  // ---- init state / outputs column 0 ----
  for (int idx = tid; idx < BB * LPP; idx += NTHR) {
    P.prob_list[idx] = 0.f; P.cand_list[idx] = 0;
  }
  for (int idx = tid; idx < BB * LNN; idx += NTHR) P.nrec[idx] = 0.f;
  if (tid < BB) {
    int b = tid;
    int   m0 = P.marker[b * TT];
    float t0 = P.timed[b * TT];
    sMark[b] = m0; sTime[b] = t0; sChosen[b] = 0;
    P.prob_list[b * LPP] = 1.f;
    P.cand_list[b * LPP] = m0;
    P.nrec[b * LNN] = 1.f;
    P.out[0 * BB * TT + b * TT] = (float)m0;
    P.out[1 * BB * TT + b * TT] = t0;
    P.out[2 * BB * TT + b * TT] = P.maskd[b * TT];
    P.out[3 * BB * TT + b * TT] = 1.f;
    P.out[4 * BB * TT + b * TT] = 1.f;
  }
  __syncthreads();

  const float scale = 0.125f;  // 1/sqrt(DK)

  for (int i = 0; i < TMM; ++i) {
    // ---- vvec = emb[lm] + BETA*(t*te_w + te_b) -> sA[:, :E] ----
    for (int idx = tid; idx < BB * EE; idx += NTHR) {
      int b = idx >> 7, e = idx & 127;
      sA[b][e] = P.emb[(size_t)sMark[b] * EE + e] +
                 0.1f * (sTime[b] * P.te_w[e] + P.te_b[e]);
    }
    __syncthreads();

    // ---- md = lrelu(vvec @ el_w + el_b) -> sB ----
    gemm32(&sA[0][0], DD, P.el_w, EE, DD, &sB[0][0], DD, tid);
    __syncthreads();
    for (int idx = tid; idx < BB * DD; idx += NTHR) {
      int n = idx & (DD - 1);
      float x = sB[0][idx] + P.el_b[n];
      sB[0][idx] = (x >= 0.f) ? x : 0.01f * x;
    }
    __syncthreads();

    // ---- q,k,v projections (k/v straight into caches) ----
    gemm32(&sB[0][0], DD, P.wq, DD, DD, P.g_q, DD, tid);
    gemm32(&sB[0][0], DD, P.wk, DD, DD, P.kc + (size_t)i * DD, TMM * DD, tid);
    gemm32(&sB[0][0], DD, P.wv, DD, DD, P.vc + (size_t)i * DD, TMM * DD, tid);
    __syncthreads();

    // ---- causal attention over cache (one wave per (b,h)) ----
    for (int pair = w; pair < BB * HH; pair += 16) {
      int b = pair >> 3, h = pair & 7;
      const float* qp = P.g_q + (size_t)b * DD + h * DKK;
      float q0 = qp[lane], q1 = qp[lane + 32];
      for (int t = 0; t <= i; ++t) {
        const float* kp = P.kc + ((size_t)b * TMM + t) * DD + h * DKK;
        float d = q0 * kp[lane] + q1 * kp[lane + 32];
        d = wred_sum(d) * scale;
        if (lane == 0) sScore[w][t] = d;
      }
      float mx = -3.0e38f;
      for (int t = lane; t <= i; t += 32) mx = fmaxf(mx, sScore[w][t]);
      mx = wred_max(mx);
      float sum = 0.f;
      for (int t = lane; t <= i; t += 32) sum += expf(sScore[w][t] - mx);
      sum = wred_sum(sum);
      float c0 = 0.f, c1 = 0.f;
      for (int t = 0; t <= i; ++t) {
        float pz = expf(sScore[w][t] - mx);
        const float* vp = P.vc + ((size_t)b * TMM + t) * DD + h * DKK;
        c0 += pz * vp[lane]; c1 += pz * vp[lane + 32];
      }
      P.g_ctx[(size_t)b * DD + h * DKK + lane]      = c0 / sum;
      P.g_ctx[(size_t)b * DD + h * DKK + lane + 32] = c1 / sum;
    }
    __syncthreads();

    // ---- x1 = LN1(md + ctx @ wo + wo_b) -> sA ----
    gemm32(P.g_ctx, DD, P.wo, DD, DD, &sA[0][0], DD, tid);
    __syncthreads();
    for (int idx = tid; idx < BB * DD; idx += NTHR) {
      int n = idx & (DD - 1);
      sA[0][idx] += sB[0][idx] + P.wo_b[n];
    }
    __syncthreads();
    ln_rows(&sA[0][0], P.ln1_s, P.ln1_b, w, lane);
    __syncthreads();

    // ---- h = relu(x1 @ ff1 + ff1_b) -> g_h ----
    gemm32(&sA[0][0], DD, P.ff1, DD, DII, P.g_h, DII, tid);
    __syncthreads();
    for (int idx = tid; idx < BB * DII; idx += NTHR) {
      int n = idx & (DII - 1);
      float x = P.g_h[idx] + P.ff1_b[n];
      P.g_h[idx] = (x > 0.f) ? x : 0.f;
    }
    __syncthreads();

    // ---- inten = LN2(x1 + h @ ff2 + ff2_b) -> sB ----
    gemm32(P.g_h, DII, P.ff2, DII, DD, &sB[0][0], DD, tid);
    __syncthreads();
    for (int idx = tid; idx < BB * DD; idx += NTHR) {
      int n = idx & (DD - 1);
      sB[0][idx] += sA[0][idx] + P.ff2_b[n];
    }
    __syncthreads();
    ln_rows(&sB[0][0], P.ln2_s, P.ln2_b, w, lane);
    __syncthreads();

    // ---- new_time = t + softplus(inten.tl_w + tl_b); intdot = inten.ml_w[E:]
    for (int r = w; r < BB; r += 16) {
      float a1 = 0.f, a2 = 0.f;
      for (int k = lane; k < DD; k += 32) {
        float x = sB[r][k];
        a1 += x * P.tl_w[k];
        a2 += x * P.ml_w[EE + k];
      }
      a1 = wred_sum(a1); a2 = wred_sum(a2);
      if (lane == 0) {
        float z = a1 + P.tl_b[0];
        float sp = (z > 20.f) ? z : log1pf(expf(z));
        sTime[r] = sTime[r] + sp;
        sDot[r] = a2;
      }
    }
    __syncthreads();

    // ---- nrec slice update ----
    for (int idx = tid; idx < BB * SS; idx += NTHR) {
      int b = idx >> 5, s = idx & 31;
      P.nrec[b * LNN + 1 + i * SS + s] =
          P.nprob[(size_t)sMark[b] * SS + s];
    }
    // ---- mw[b][s] = emb[nl[s]].ml_w[:E] + intdot[b] + ml_b ----
    for (int idx = tid; idx < BB * SS; idx += NTHR) {
      int b = idx >> 5, s = idx & 31;
      int nb = P.nlist[(size_t)sMark[b] * SS + s];
      const float* er = P.emb + (size_t)nb * EE;
      float acc = P.ml_b[0] + sDot[b];
      for (int e = 0; e < EE; ++e) acc += er[e] * P.ml_w[e];
      sMW[b][s] = acc;
    }
    __syncthreads();
    // softmax rows of sMW (one wave per row)
    for (int r = w; r < BB; r += 16) {
      float v = sMW[r][lane];
      float mx = wred_max(v);
      float e = expf(v - mx);
      float sm = wred_sum(e);
      sMW[r][lane] = e / sm;
    }
    __syncthreads();

    // ---- probability-tree update (serial per b) ----
    if (tid < BB) {
      int b = tid;
      int ch = sChosen[b];
      float base = P.prob_list[b * LPP + ch];
      const int* nl = P.nlist + (size_t)sMark[b] * SS;
      P.prob_list[b * LPP + ch] = base * sMW[b][0];
      for (int s = 1; s < SS; ++s) {
        P.prob_list[b * LPP + 1 + i * (SS - 1) + (s - 1)] = base * sMW[b][s];
        P.cand_list[b * LPP + 1 + i * (SS - 1) + (s - 1)] = nl[s];
      }
    }
    __syncthreads();

    // ---- categorical sampling: argmax(log p + gumbel), wave per b ----
    for (int r = w; r < BB; r += 16) {
      float best = -3.0e38f; int bi = 0;
      for (int idx = lane; idx < LPP; idx += 32) {
        float pl = P.prob_list[r * LPP + idx];
        float sc = logf(pl) +
                   gumbelf(((unsigned)i * 1000003u) ^
                           ((unsigned)r * 8191u) ^ (unsigned)idx);
        if (sc > best) { best = sc; bi = idx; }
      }
#pragma unroll
      for (int o = 16; o; o >>= 1) {
        float ov = __shfl_xor(best, o, 32);
        int   oi = __shfl_xor(bi, o, 32);
        if (ov > best || (ov == best && oi < bi)) { best = ov; bi = oi; }
      }
      if (lane == 0) {
        sChosen[r] = bi;
        int nm = P.cand_list[r * LPP + bi];
        sMark[r] = nm;
        float nt = sTime[r];
        P.out[0 * BB * TT + r * TT + i + 1] = (float)nm;
        P.out[1 * BB * TT + r * TT + i + 1] = nt;
        P.out[2 * BB * TT + r * TT + i + 1] = (nt < 50.0f) ? 1.f : 0.f;
        P.out[3 * BB * TT + r * TT + i + 1] = P.nrec[r * LNN + bi];
        P.out[4 * BB * TT + r * TT + i + 1] = P.prob_list[r * LPP + bi];
      }
    }
    __syncthreads();
  }
}

// ---------------- host launcher ----------------
extern "C" void kernel_launch(void* const* d_in, const int* in_sizes, int n_in,
                              void* d_out, int out_size, void* d_ws, size_t ws_size,
                              hipStream_t stream) {
  (void)in_sizes; (void)n_in; (void)out_size; (void)ws_size;
  const int*   marker = (const int*)  d_in[0];
  const float* timed  = (const float*)d_in[1];
  const float* maskd  = (const float*)d_in[2];
  const float* emb    = (const float*)d_in[3];
  const float* te_w   = (const float*)d_in[4];
  const float* te_b   = (const float*)d_in[5];
  const float* el_w   = (const float*)d_in[6];
  const float* el_b   = (const float*)d_in[7];
  const float* s1_w   = (const float*)d_in[8];
  const float* s1_b   = (const float*)d_in[9];
  const float* s2_w   = (const float*)d_in[10];
  const float* s2_b   = (const float*)d_in[11];
  const float* ml_w   = (const float*)d_in[12];
  const float* ml_b   = (const float*)d_in[13];
  const float* tl_w   = (const float*)d_in[14];
  const float* tl_b   = (const float*)d_in[15];
  const float* wq     = (const float*)d_in[16];
  const float* wk     = (const float*)d_in[17];
  const float* wv     = (const float*)d_in[18];
  const float* wo     = (const float*)d_in[19];
  const float* wo_b   = (const float*)d_in[20];
  const float* ln1_s  = (const float*)d_in[21];
  const float* ln1_b  = (const float*)d_in[22];
  const float* ff1_w  = (const float*)d_in[23];
  const float* ff1_b  = (const float*)d_in[24];
  const float* ff2_w  = (const float*)d_in[25];
  const float* ff2_b  = (const float*)d_in[26];
  const float* ln2_s  = (const float*)d_in[27];
  const float* ln2_b  = (const float*)d_in[28];

  // workspace layout
  char* ws = (char*)d_ws;
  size_t off = 0;
  auto take = [&](size_t bytes) -> char* {
    char* p = ws + off;
    off = (off + bytes + 255) & ~(size_t)255;
    return p;
  };
  __bf16* el_bf  = (__bf16*)take((size_t)EE * DD * 2);
  __bf16* wq_bf  = (__bf16*)take((size_t)DD * DD * 2);
  __bf16* wk_bf  = (__bf16*)take((size_t)DD * DD * 2);
  __bf16* wv_bf  = (__bf16*)take((size_t)DD * DD * 2);
  __bf16* wo_bf  = (__bf16*)take((size_t)DD * DD * 2);
  __bf16* ff1_bf = (__bf16*)take((size_t)DD * DII * 2);
  __bf16* ff2_bf = (__bf16*)take((size_t)DII * DD * 2);
  __bf16* s1a_bf = (__bf16*)take((size_t)EE * EE * 2);
  __bf16* s1b_bf = (__bf16*)take((size_t)EE * EE * 2);
  int*    cand   = (int*)   take((size_t)CC * 4);
  float*  Acand  = (float*) take((size_t)CC * EE * 4);
  float*  Bm     = (float*) take((size_t)MM * EE * 4);
  float*  pbuf   = (float*) take((size_t)MM * CC * 4);
  int*    nlist  = (int*)   take((size_t)MM * SS * 4);
  float*  nprob  = (float*) take((size_t)MM * SS * 4);
  float*  kc     = (float*) take((size_t)BB * TMM * DD * 4);
  float*  vc     = (float*) take((size_t)BB * TMM * DD * 4);
  float*  g_q    = (float*) take((size_t)BB * DD * 4);
  float*  g_ctx  = (float*) take((size_t)BB * DD * 4);
  float*  g_h    = (float*) take((size_t)BB * DII * 4);
  float*  probl  = (float*) take((size_t)BB * LPP * 4);
  int*    candl  = (int*)   take((size_t)BB * LPP * 4);
  float*  nrec   = (float*) take((size_t)BB * LNN * 4);

  // 1) weights -> bf16, pre-swizzled into WMMA B-fragment layout
  k_swizzle_w<<<256, 256, 0, stream>>>(el_w,  el_bf,  EE,  DD);
  k_swizzle_w<<<256, 256, 0, stream>>>(wq,    wq_bf,  DD,  DD);
  k_swizzle_w<<<256, 256, 0, stream>>>(wk,    wk_bf,  DD,  DD);
  k_swizzle_w<<<256, 256, 0, stream>>>(wv,    wv_bf,  DD,  DD);
  k_swizzle_w<<<256, 256, 0, stream>>>(wo,    wo_bf,  DD,  DD);
  k_swizzle_w<<<512, 256, 0, stream>>>(ff1_w, ff1_bf, DD,  DII);
  k_swizzle_w<<<512, 256, 0, stream>>>(ff2_w, ff2_bf, DII, DD);
  k_swizzle_w<<<64,  256, 0, stream>>>(s1_w,           s1a_bf, EE, EE);
  k_swizzle_w<<<64,  256, 0, stream>>>(s1_w + EE * EE, s1b_bf, EE, EE);

  // 2) neighbor precompute
  k_cand_init<<<1, CC, 0, stream>>>(cand);
  {
    int tiles = ((CC + 15) / 16) * (EE / 16);          // 128
    k_gemm_pre<<<(tiles + 3) / 4, 128, 0, stream>>>(emb, cand, CC, EE,
                                                    (const v16bf*)s1a_bf, EE, Acand);
  }
  {
    int tiles = ((MM + 15) / 16) * (EE / 16);          // 2504
    k_gemm_pre<<<(tiles + 3) / 4, 128, 0, stream>>>(emb, nullptr, MM, EE,
                                                    (const v16bf*)s1b_bf, EE, Bm);
  }
  k_pscore<<<MM, CC, 0, stream>>>(Acand, Bm, s1_b, s2_w, s2_b, pbuf);
  k_topk<<<MM, 32, 0, stream>>>(pbuf, cand, nlist, nprob);

  // 3) persistent sequential transformer scan (one WGP, 16 wave32)
  MainParams P;
  P.marker = marker; P.timed = timed; P.maskd = maskd;
  P.emb = emb; P.te_w = te_w; P.te_b = te_b; P.el_b = el_b;
  P.wo_b = wo_b; P.ln1_s = ln1_s; P.ln1_b = ln1_b;
  P.ff1_b = ff1_b; P.ff2_b = ff2_b; P.ln2_s = ln2_s; P.ln2_b = ln2_b;
  P.tl_w = tl_w; P.tl_b = tl_b; P.ml_w = ml_w; P.ml_b = ml_b;
  P.el_w = (const v16bf*)el_bf; P.wq = (const v16bf*)wq_bf;
  P.wk = (const v16bf*)wk_bf;   P.wv = (const v16bf*)wv_bf;
  P.wo = (const v16bf*)wo_bf;   P.ff1 = (const v16bf*)ff1_bf;
  P.ff2 = (const v16bf*)ff2_bf;
  P.nlist = nlist; P.nprob = nprob;
  P.kc = kc; P.vc = vc; P.g_q = g_q; P.g_ctx = g_ctx; P.g_h = g_h;
  P.prob_list = probl; P.cand_list = candl; P.nrec = nrec;
  P.out = (float*)d_out;
  k_main<<<1, NTHR, 0, stream>>>(P);
}
